// TernaryLinear_22582938042540
// MI455X (gfx1250) — matrix operand order repaired and checked by reading, not compiled, not run on hardware
//
#include <hip/hip_runtime.h>
#include <stdint.h>

#ifndef __has_builtin
#define __has_builtin(x) 0
#endif

typedef __attribute__((ext_vector_type(16))) __bf16 v16bf;
typedef __attribute__((ext_vector_type(8)))  __bf16 v8bf;
typedef __attribute__((ext_vector_type(8)))  float  v8f;
typedef __attribute__((ext_vector_type(4)))  float  v4f;
typedef __attribute__((ext_vector_type(4)))  int    v4i;

// Async global->LDS (gfx1250): probe-verified signature
// (v4i addrspace(1)* src, v4i addrspace(3)* dst, imm offset, imm cpol)
#if defined(__HIP_DEVICE_COMPILE__) && __has_builtin(__builtin_amdgcn_global_load_async_to_lds_b128)
#define HAVE_ASYNC 1
typedef __attribute__((address_space(1))) v4i* g_v4i_p;
typedef __attribute__((address_space(3))) v4i* l_v4i_p;
#else
#define HAVE_ASYNC 0
#endif

#define IN_F   4096
#define OUT_F  4096
#define N_TOK  8192
#define RANK   32

#define BM 128
#define BN 256
#define BK 32
#define NKSTEP (IN_F / BK)   // 128

// bf16 LDS tiles: 32 halves/row padded to 40 (80 B stride; 16B-aligned,
// bank step 20 mod 64 -> 16 lanes hit distinct banks)
#define SROW 40

static __device__ __forceinline__ void wait_async_zero() {
#if HAVE_ASYNC
#if __has_builtin(__builtin_amdgcn_s_wait_asynccnt)
  __builtin_amdgcn_s_wait_asynccnt(0);
#else
  asm volatile("s_wait_asynccnt 0" ::: "memory");
#endif
#endif
}

static __device__ __forceinline__ v8f wmma_bf16(v16bf a, v16bf b, v8f c) {
  return __builtin_amdgcn_wmma_f32_16x16x32_bf16(
      false, a, false, b, (short)0, c, false, false);
}

// 16 f32 -> 16 bf16 (two 8-chunks, canonical fragment ordering)
static __device__ __forceinline__ v16bf cvt16_2x8(const float* p0, const float* p1) {
  v4f f0 = *(const v4f*)p0;
  v4f f1 = *(const v4f*)(p0 + 4);
  v4f f2 = *(const v4f*)p1;
  v4f f3 = *(const v4f*)(p1 + 4);
  v16bf r;
#pragma unroll
  for (int j = 0; j < 4; ++j) {
    r[j]      = (__bf16)f0[j];
    r[4 + j]  = (__bf16)f1[j];
    r[8 + j]  = (__bf16)f2[j];
    r[12 + j] = (__bf16)f3[j];
  }
  return r;
}
static __device__ __forceinline__ v16bf cvt16_c(const float* p) { return cvt16_2x8(p, p + 8); }

// assemble v16bf from two 16B LDS chunks
static __device__ __forceinline__ v16bf frag16(const __bf16* p0, const __bf16* p1) {
  v8bf lo = *(const v8bf*)p0;
  v8bf hi = *(const v8bf*)p1;
  v16bf r;
#pragma unroll
  for (int j = 0; j < 8; ++j) { r[j] = lo[j]; r[8 + j] = hi[j]; }
  return r;
}

// ---------------------------------------------------------------------------
// Kernel A: T[t, r] = sum_i x[t,i] * lora_B[r,i]   (T: [N_TOK, RANK] f32 in ws)
// ---------------------------------------------------------------------------
__global__ __launch_bounds__(256) void lora_xB_kernel(
    const float* __restrict__ x, const float* __restrict__ loraB,
    float* __restrict__ T) {
  const int tid  = threadIdx.x;
  const int wave = tid >> 5, lane = tid & 31;
  const int l16 = lane & 15, hg = lane >> 4;
  const int s8  = hg * 8;
  const int trow = blockIdx.x * 128 + wave * 16;

  v8f acc0 = {}, acc1 = {};
  const float* xrow  = x + (size_t)(trow + l16) * IN_F;
  const float* b0row = loraB + (size_t)l16 * IN_F + s8;
  const float* b1row = loraB + (size_t)(16 + l16) * IN_F + s8;

  for (int k0 = 0; k0 < IN_F; k0 += BK) {
    v16bf a  = cvt16_2x8(xrow + k0 + s8, xrow + k0 + 16 + s8);
    v16bf b0 = cvt16_2x8(b0row + k0, b0row + k0 + 16);
    v16bf b1 = cvt16_2x8(b1row + k0, b1row + k0 + 16);
    acc0 = wmma_bf16(a, b0, acc0);
    acc1 = wmma_bf16(a, b1, acc1);
  }
  float* tp = T + (size_t)(trow + hg * 8) * RANK + l16;
#pragma unroll
  for (int j = 0; j < 8; ++j) {
    tp[(size_t)j * RANK]      = acc0[j];
    tp[(size_t)j * RANK + 16] = acc1[j];
  }
}

// ---------------------------------------------------------------------------
// W dequant precompute: wq int8 {-1,0,1} -> bf16 matrix in ws (memory-bound)
// ---------------------------------------------------------------------------
__global__ __launch_bounds__(256) void wq_bf16_kernel(
    const signed char* __restrict__ wq, __bf16* __restrict__ wb) {
  size_t base = ((size_t)blockIdx.x * 256 + threadIdx.x) * 16;
  union { v4i v; signed char c[16]; } u;
  u.v = *(const v4i*)(wq + base);
  v8bf h0, h1;
#pragma unroll
  for (int i = 0; i < 8; ++i) {
    h0[i] = (__bf16)(float)u.c[i];
    h1[i] = (__bf16)(float)u.c[8 + i];
  }
  *(v8bf*)(wb + base)     = h0;
  *(v8bf*)(wb + base + 8) = h1;
}

// ---------------------------------------------------------------------------
// Kernel B: out = (x @ Wq^T)*scale + T @ A^T + bias
// Block 128x256, 8 waves (2x4), wave tile 64x64 (16 accumulators).
// PREW=true : W streamed as precomputed bf16 via async global->LDS (no VALU)
// PREW=false: W converted int8->bf16 in-loop (ws too small fallback)
// ---------------------------------------------------------------------------
template <bool PREW>
__global__ __launch_bounds__(256) void ternary_main_kernel(
    const float* __restrict__ x, const signed char* __restrict__ wq,
    const __bf16* __restrict__ wbf,
    const float* __restrict__ scale, const float* __restrict__ loraA,
    const float* __restrict__ bias, const float* __restrict__ T,
    float* __restrict__ out) {
  __shared__ __bf16 sX[2][BM * SROW];   // 20 KB
  __shared__ __bf16 sW[2][BN * SROW];   // 40 KB

  const int tid    = threadIdx.x;
  const int wave   = tid >> 5, lane = tid & 31;
  const int wave_m = wave >> 2, wave_n = wave & 3;   // 2 x 4 wave grid
  const int l16 = lane & 15, hg = lane >> 4;
  const int s8  = hg * 8;

  const int tbase = blockIdx.y * BM;
  const int nbase = blockIdx.x * BN;

  // x staging: 128 rows x 2 half-rows(16 f32) -> one half-row per thread
  const int xrow = tid >> 1;
  const int xh   = (tid & 1) << 4;
  const float* xgp = x + (size_t)(tbase + xrow) * IN_F + xh;

  struct XR { v4f f[4]; };
  auto loadX = [&](int k0) -> XR {
    XR r;
#pragma unroll
    for (int i = 0; i < 4; ++i) r.f[i] = *(const v4f*)(xgp + k0 + i * 4);
    return r;
  };
  auto storeX = [&](const XR& r, int buf) {
    v8bf h0, h1;
#pragma unroll
    for (int j = 0; j < 4; ++j) {
      h0[j]     = (__bf16)r.f[0][j];
      h0[4 + j] = (__bf16)r.f[1][j];
      h1[j]     = (__bf16)r.f[2][j];
      h1[4 + j] = (__bf16)r.f[3][j];
    }
    __bf16* lp = &sX[buf][xrow * SROW + xh];
    *(v8bf*)lp       = h0;
    *(v8bf*)(lp + 8) = h1;
  };
  auto stageW_async = [&](int k0, int buf) {
    // 256 rows x 4 chunks(16B bf16) = 1024 chunks -> 4 per thread
#pragma unroll
    for (int i = 0; i < 4; ++i) {
      int chunk = tid + i * 256;
      int row = chunk >> 2;
      int c8  = (chunk & 3) << 3;  // halves
      const __bf16* gp = wbf + (size_t)(nbase + row) * IN_F + k0 + c8;
      __bf16* lp = &sW[buf][row * SROW + c8];
#if HAVE_ASYNC
      __builtin_amdgcn_global_load_async_to_lds_b128((g_v4i_p)gp, (l_v4i_p)lp, 0, 0);
#else
      *(v8bf*)lp = *(const v8bf*)gp;
#endif
    }
  };
  // !PREW: 256 rows x 2 chunks(16 int8) = 512 chunks -> 2 per thread
  auto loadW = [&](int k0, v4i wr[2]) {
#pragma unroll
    for (int i = 0; i < 2; ++i) {
      int chunk = tid + i * 256;
      int row = chunk >> 1, c = (chunk & 1) << 4;
      wr[i] = *(const v4i*)(wq + (size_t)(nbase + row) * IN_F + k0 + c);
    }
  };
  auto storeW = [&](const v4i wr[2], int buf) {
#pragma unroll
    for (int i = 0; i < 2; ++i) {
      int chunk = tid + i * 256;
      int row = chunk >> 1, c = (chunk & 1) << 4;
      union { v4i v; signed char ch[16]; } u; u.v = wr[i];
      v8bf h0, h1;
#pragma unroll
      for (int j = 0; j < 8; ++j) {
        h0[j] = (__bf16)(float)u.ch[j];
        h1[j] = (__bf16)(float)u.ch[8 + j];
      }
      __bf16* lp = &sW[buf][row * SROW + c];
      *(v8bf*)lp       = h0;
      *(v8bf*)(lp + 8) = h1;
    }
  };

  // ---- prologue: fill buffer 0 ----
  {
    v4i w0[2];
    if (PREW) stageW_async(0, 0); else loadW(0, w0);
    XR xr = loadX(0);
    storeX(xr, 0);
    if (!PREW) storeW(w0, 0);
    wait_async_zero();
    __syncthreads();
  }

  v8f zero = {};
  v8f acc[4][4];
#pragma unroll
  for (int mt = 0; mt < 4; ++mt)
#pragma unroll
    for (int nt = 0; nt < 4; ++nt) acc[mt][nt] = zero;

  // ---- main K loop, 1-deep software pipeline ----
  for (int ks = 0; ks < NKSTEP; ++ks) {
    const int cur = ks & 1, nxt = cur ^ 1;
    const bool more = (ks + 1 < NKSTEP);
    XR  xr;
    v4i wr[2];
    if (more) {
      if (PREW) stageW_async((ks + 1) * BK, nxt);  // async copies fly over compute
      else      loadW((ks + 1) * BK, wr);
      xr = loadX((ks + 1) * BK);
    }

    v16bf bf[4];
#pragma unroll
    for (int nt = 0; nt < 4; ++nt) {
      const __bf16* p = &sW[cur][(wave_n * 64 + nt * 16 + l16) * SROW];
      bf[nt] = frag16(p + s8, p + 16 + s8);
    }
#pragma unroll
    for (int mt = 0; mt < 4; ++mt) {
      const __bf16* p = &sX[cur][(wave_m * 64 + mt * 16 + l16) * SROW];
      v16bf af = frag16(p + s8, p + 16 + s8);
#pragma unroll
      for (int nt = 0; nt < 4; ++nt)
        acc[mt][nt] = wmma_bf16(af, bf[nt], acc[mt][nt]);
    }

    if (more) {
      storeX(xr, nxt);
      if (!PREW) storeW(wr, nxt);
    }
    wait_async_zero();
    __syncthreads();
  }

  // ---- epilogue: D = T x A^T + (acc*scale + bias), one WMMA per acc ----
  float sc[4], bi[4];
  v16bf afrag[4];
#pragma unroll
  for (int nt = 0; nt < 4; ++nt) {
    int o = nbase + wave_n * 64 + nt * 16 + l16;
    sc[nt] = scale[o];
    bi[nt] = bias[o];
    afrag[nt] = cvt16_2x8(loraA + (size_t)o * RANK + s8, loraA + (size_t)o * RANK + 16 + s8);
  }
#pragma unroll
  for (int mt = 0; mt < 4; ++mt) {
    const float* tp = T + (size_t)(tbase + wave_m * 64 + mt * 16 + l16) * RANK;
    v16bf tf = cvt16_2x8(tp + s8, tp + 16 + s8);
#pragma unroll
    for (int nt = 0; nt < 4; ++nt) {
      v8f c;
#pragma unroll
      for (int j = 0; j < 8; ++j) c[j] = acc[mt][nt][j] * sc[nt] + bi[nt];
      v8f d = wmma_bf16(tf, afrag[nt], c);
      int o = nbase + wave_n * 64 + nt * 16 + l16;
      float* op = out + (size_t)(tbase + wave_m * 64 + mt * 16 + hg * 8) * OUT_F + o;
#pragma unroll
      for (int j = 0; j < 8; ++j)
        __builtin_nontemporal_store(d[j], op + (size_t)j * OUT_F);
    }
  }
}

extern "C" void kernel_launch(void* const* d_in, const int* in_sizes, int n_in,
                              void* d_out, int out_size, void* d_ws, size_t ws_size,
                              hipStream_t stream) {
  (void)in_sizes; (void)n_in; (void)out_size;
  const float*       x     = (const float*)d_in[0];
  const signed char* wq    = (const signed char*)d_in[1];
  const float*       scale = (const float*)d_in[2];
  const float*       loraA = (const float*)d_in[3];
  const float*       loraB = (const float*)d_in[4];
  const float*       bias  = (const float*)d_in[5];
  float* out = (float*)d_out;

  float* T = (float*)d_ws;                               // 1 MB
  const size_t offW  = (size_t)N_TOK * RANK * sizeof(float);
  const size_t needW = (size_t)OUT_F * IN_F * 2;         // 32 MB bf16 W
  __bf16* wb = (__bf16*)((char*)d_ws + offW);
  const bool prew = ws_size >= offW + needW;             // fixed per harness -> deterministic

  lora_xB_kernel<<<dim3(N_TOK / 128), dim3(256), 0, stream>>>(x, loraB, T);
  dim3 grid(OUT_F / BN, N_TOK / BM);
  if (prew) {
    wq_bf16_kernel<<<dim3((OUT_F * (size_t)IN_F) / (256 * 16)), dim3(256), 0, stream>>>(wq, wb);
    ternary_main_kernel<true><<<grid, dim3(256), 0, stream>>>(
        x, wq, wb, scale, loraA, bias, T, out);
  } else {
    ternary_main_kernel<false><<<grid, dim3(256), 0, stream>>>(
        x, wq, nullptr, scale, loraA, bias, T, out);
  }
}
